// Decoder_86423331930695
// MI455X (gfx1250) — matrix operand order repaired and checked by reading, not compiled, not run on hardware
//
#include <hip/hip_runtime.h>

#define DEV __device__ __forceinline__

typedef __attribute__((ext_vector_type(16))) __bf16 v16bf;
typedef __attribute__((ext_vector_type(8)))  float  v8f;
typedef __attribute__((ext_vector_type(4)))  float  f32x4;
typedef __attribute__((ext_vector_type(8)))  unsigned short ushort8;

struct BFrag { ushort8 lo, hi; };

DEV v16bf frag_cast(BFrag f) { return __builtin_bit_cast(v16bf, f); }

DEV unsigned short f2bf(float f) {
  unsigned u = __builtin_bit_cast(unsigned, f);
  u += 0x7FFFu + ((u >> 16) & 1u);
  return (unsigned short)(u >> 16);
}
DEV float sigm(float x) { return 1.f / (1.f + __expf(-x)); }

// ---------------------------------------------------------------------------
// M-blocked bf16 WMMA GEMM for M==64 (4 M-tiles per wave):
//   C[64,N] (+)= A[64,K] * Bw[N,K]^T (+ bias0 + bias1)
// Each wave owns one 16-wide N tile and all 4 M tiles: every B fragment is
// loaded once and fed to 4 WMMAs -> 5/8 the load bytes per WMMA vs 1-tile.
// ---------------------------------------------------------------------------
__global__ void gemm_bf16_wmma_m4(const unsigned short* __restrict__ A, int lda,
                                  const unsigned short* __restrict__ Bw, int ldb,
                                  float* __restrict__ C, int ldc,
                                  const float* __restrict__ bias0,
                                  const float* __restrict__ bias1,
                                  int Ntiles, int K, int beta) {
  const int lane = threadIdx.x & 31;
  const int w = blockIdx.x * (blockDim.x >> 5) + (threadIdx.x >> 5);
  if (w >= Ntiles) return;                 // wave-uniform: EXEC stays all-ones
  const int half = lane >> 4;              // 0: lanes 0-15, 1: lanes 16-31
  const int l15 = lane & 15;
  const int n = w * 16 + l15;

  v8f acc[4];
  if (beta) {
    for (int mt = 0; mt < 4; ++mt)
      for (int r = 0; r < 8; ++r)
        acc[mt][r] = C[(size_t)(mt * 16 + (half ? 8 : 0) + r) * ldc + n];
  } else {
    for (int mt = 0; mt < 4; ++mt)
      for (int r = 0; r < 8; ++r) acc[mt][r] = 0.f;
  }

  // A 16x32 bf16: lanes 0-15 -> K 0..15; lanes 16-31 -> K 16..31
  const unsigned short* ap = A + (size_t)l15 * lda + (half ? 16 : 0);
  // B 32x16 bf16: lanes 0-15 -> K 0..15; lanes 16-31 -> K 16..31; N = l15
  const unsigned short* bp = Bw + (size_t)n * ldb + (half ? 16 : 0);

  for (int kk = 0; kk < K; kk += 32) {
    __builtin_prefetch(bp + kk + 256, 0, 1);   // global_prefetch_b8, ~512B ahead
    BFrag bfr;
    bfr.lo = *(const ushort8*)(bp + kk);
    bfr.hi = *(const ushort8*)(bp + kk + 8);
    const v16bf bv = frag_cast(bfr);
#pragma unroll
    for (int mt = 0; mt < 4; ++mt) {
      const unsigned short* amt = ap + (size_t)(mt * 16) * lda;
      BFrag af;
      af.lo = *(const ushort8*)(amt + kk);
      af.hi = *(const ushort8*)(amt + kk + 8);
      acc[mt] = __builtin_amdgcn_wmma_f32_16x16x32_bf16(false, frag_cast(af),
                                                        false, bv,
                                                        (short)0, acc[mt],
                                                        false, false);
    }
  }

  float bs = 0.f;
  if (bias0) bs += bias0[n];
  if (bias1) bs += bias1[n];
  for (int mt = 0; mt < 4; ++mt)
    for (int r = 0; r < 8; ++r)
      C[(size_t)(mt * 16 + (half ? 8 : 0) + r) * ldc + n] = acc[mt][r] + bs;
}

// ---------------------------------------------------------------------------
// Generic 1-tile-per-wave WMMA GEMM (used for the large-M wh precompute).
// ---------------------------------------------------------------------------
__global__ void gemm_bf16_wmma(const unsigned short* __restrict__ A, int lda,
                               const unsigned short* __restrict__ Bw, int ldb,
                               float* __restrict__ C, int ldc,
                               int Mtiles, int Ntiles, int K) {
  const int lane = threadIdx.x & 31;
  const int w = blockIdx.x * (blockDim.x >> 5) + (threadIdx.x >> 5);
  if (w >= Mtiles * Ntiles) return;
  const int tm = w % Mtiles, tn = w / Mtiles;
  const int half = lane >> 4;
  const int l15 = lane & 15;
  const int n  = tn * 16 + l15;
  const int mb = tm * 16 + (half ? 8 : 0);

  v8f acc;
  for (int r = 0; r < 8; ++r) acc[r] = 0.f;

  const unsigned short* ap = A + (size_t)(tm * 16 + l15) * lda + (half ? 16 : 0);
  const unsigned short* bp = Bw + (size_t)n * ldb + (half ? 16 : 0);

  for (int kk = 0; kk < K; kk += 32) {
    BFrag af, bf;
    af.lo = *(const ushort8*)(ap + kk);
    af.hi = *(const ushort8*)(ap + kk + 8);
    bf.lo = *(const ushort8*)(bp + kk);
    bf.hi = *(const ushort8*)(bp + kk + 8);
    acc = __builtin_amdgcn_wmma_f32_16x16x32_bf16(false, frag_cast(af),
                                                  false, frag_cast(bf),
                                                  (short)0, acc, false, false);
  }
  for (int r = 0; r < 8; ++r) C[(size_t)(mb + r) * ldc + n] = acc[r];
}

// ---------------------------------------------------------------------------
__global__ void f32_to_bf16_vec8(const float* __restrict__ src,
                                 unsigned short* __restrict__ dst, size_t n8) {
  size_t i = (size_t)blockIdx.x * blockDim.x + threadIdx.x;
  if (i >= n8) return;
  const f32x4* s = (const f32x4*)src + i * 2;
  f32x4 a = s[0], b = s[1];
  ushort8 o;
  o[0] = f2bf(a[0]); o[1] = f2bf(a[1]); o[2] = f2bf(a[2]); o[3] = f2bf(a[3]);
  o[4] = f2bf(b[0]); o[5] = f2bf(b[1]); o[6] = f2bf(b[2]); o[7] = f2bf(b[3]);
  ((ushort8*)dst)[i] = o;
}

__global__ void init_state_kernel(const float* __restrict__ h0, const float* __restrict__ c0,
                                  float* __restrict__ h, float* __restrict__ c,
                                  unsigned short* __restrict__ hbf, float* __restrict__ lossacc) {
  int idx = blockIdx.x * blockDim.x + threadIdx.x;
  if (idx < 64) lossacc[idx] = 0.f;
  if (idx >= 64 * 1024) return;
  float hv = h0[idx];
  h[idx] = hv; c[idx] = c0[idx]; hbf[idx] = f2bf(hv);
}

__global__ void embed_kernel(const int* __restrict__ labels, const float* __restrict__ embed,
                             int t, float* __restrict__ x, unsigned short* __restrict__ xbf) {
  int idx = blockIdx.x * blockDim.x + threadIdx.x;
  if (idx >= 64 * 512) return;
  int b = idx >> 9, e = idx & 511;
  int tok = (t == 0) ? 1 : labels[b * 25 + (t - 1)];   // SOS=1 at t=0
  float v = embed[(size_t)tok * 512 + e];
  x[idx] = v; xbf[idx] = f2bf(v);
}

__global__ void lstm_ew_kernel(const float* __restrict__ gates, float* __restrict__ c,
                               float* __restrict__ h, unsigned short* __restrict__ hbf,
                               unsigned short* __restrict__ fcA) {
  int idx = blockIdx.x * blockDim.x + threadIdx.x;
  if (idx >= 64 * 1024) return;
  int b = idx >> 10, j = idx & 1023;
  const float* g = gates + (size_t)b * 4096;
  float ig = sigm(g[j]);
  float fg = sigm(g[1024 + j]);
  float gg = tanhf(g[2048 + j]);
  float og = sigm(g[3072 + j]);
  float c2 = fg * c[idx] + ig * gg;
  float h2 = og * tanhf(c2);
  c[idx] = c2; h[idx] = h2;
  unsigned short hb = f2bf(h2);
  hbf[idx] = hb;
  fcA[(size_t)b * 2048 + j] = hb;   // first half of [h2, ctx] FC input
}

// One block per batch row: e = tanh(wh+ws) @ v_t; softmax over S; ctx; copy gate
__global__ void attention_kernel(const float* __restrict__ wh, const float* __restrict__ wsm,
                                 const float* __restrict__ vt, const float* __restrict__ enc,
                                 const float* __restrict__ h2, const float* __restrict__ x,
                                 const float* __restrict__ w_ph, const float* __restrict__ w_ps,
                                 const float* __restrict__ w_px, const float* __restrict__ b_px,
                                 float* __restrict__ attw, float* __restrict__ ctx,
                                 unsigned short* __restrict__ fcA, float* __restrict__ ifc) {
  const int b = blockIdx.x, tid = threadIdx.x;
  const int lane = tid & 31, wv = tid >> 5;
  __shared__ float s_ws[512];
  __shared__ float s_e[128];
  __shared__ float s_red[256];

  for (int i = tid; i < 512; i += 256) s_ws[i] = wsm[b * 512 + i];
  __syncthreads();

  for (int s = wv; s < 128; s += 8) {
    const float* whrow = wh + (size_t)(b * 128 + s) * 512;
    float d = 0.f;
    for (int a = lane; a < 512; a += 32) d += tanhf(whrow[a] + s_ws[a]) * vt[a];
    for (int off = 16; off; off >>= 1) d += __shfl_xor(d, off, 32);
    if (lane == 0) s_e[s] = d;
  }
  __syncthreads();

  float ev = (tid < 128) ? s_e[tid] : -3.4e38f;
  s_red[tid] = ev; __syncthreads();
  for (int off = 128; off; off >>= 1) { if (tid < off) s_red[tid] = fmaxf(s_red[tid], s_red[tid + off]); __syncthreads(); }
  float emax = s_red[0]; __syncthreads();
  float ex = (tid < 128) ? __expf(ev - emax) : 0.f;
  s_red[tid] = ex; __syncthreads();
  for (int off = 128; off; off >>= 1) { if (tid < off) s_red[tid] += s_red[tid + off]; __syncthreads(); }
  float inv = 1.f / s_red[0]; __syncthreads();
  if (tid < 128) { float a = ex * inv; s_e[tid] = a; attw[b * 128 + tid] = a; }
  __syncthreads();

  float part = 0.f;
  for (int hh = tid; hh < 1024; hh += 256) {
    float acc = 0.f;
    const float* ecol = enc + (size_t)(b * 128) * 1024 + hh;
    for (int s = 0; s < 128; ++s) acc += s_e[s] * ecol[(size_t)s * 1024];
    ctx[b * 1024 + hh] = acc;
    fcA[(size_t)b * 2048 + 1024 + hh] = f2bf(acc);
    part += acc * w_ph[hh] + h2[b * 1024 + hh] * w_ps[hh];
  }
  for (int j = tid; j < 512; j += 256) part += x[b * 512 + j] * w_px[j];
  s_red[tid] = part; __syncthreads();
  for (int off = 128; off; off >>= 1) { if (tid < off) s_red[tid] += s_red[tid + off]; __syncthreads(); }
  if (tid == 0) ifc[b] = 1.f / (1.f + __expf(-(s_red[0] + b_px[0])));
}

// One block per batch row: softmax over V, mix with copy distribution, loss+argmax
__global__ void score_kernel(const float* __restrict__ logits, const float* __restrict__ attw,
                             const float* __restrict__ ifc, const int* __restrict__ inputs,
                             const int* __restrict__ labels, int t,
                             float* __restrict__ score, float* __restrict__ lossacc,
                             float* __restrict__ out) {
  const int b = blockIdx.x, tid = threadIdx.x;
  __shared__ float s_v[256];
  __shared__ int   s_i[256];
  const float* lrow = logits + (size_t)b * 32000;
  float* srow = score + (size_t)b * 32000;

  float m = -3.4e38f;
  for (int v = tid; v < 32000; v += 256) m = fmaxf(m, lrow[v]);
  s_v[tid] = m; __syncthreads();
  for (int off = 128; off; off >>= 1) { if (tid < off) s_v[tid] = fmaxf(s_v[tid], s_v[tid + off]); __syncthreads(); }
  const float mx = s_v[0]; __syncthreads();

  float sum = 0.f;
  for (int v = tid; v < 32000; v += 256) {      // fused: store exp, accumulate sum
    float e = __expf(lrow[v] - mx);
    srow[v] = e;
    sum += e;
  }
  s_v[tid] = sum; __syncthreads();
  for (int off = 128; off; off >>= 1) { if (tid < off) s_v[tid] += s_v[tid + off]; __syncthreads(); }
  const float fc = ifc[b];
  const float fac = (1.f - fc) / s_v[0];
  __syncthreads();

  for (int v = tid; v < 32000; v += 256) srow[v] *= fac;
  __syncthreads();
  if (tid < 128) atomicAdd(&srow[inputs[b * 128 + tid]], fc * attw[b * 128 + tid]);
  __threadfence(); __syncthreads();

  float bv = -1.f; int bi = 0;
  for (int v = tid; v < 32000; v += 256) {
    float sv = fminf(fmaxf(srow[v], 1e-30f), 1.f);
    if (sv > bv) { bv = sv; bi = v; }
  }
  s_v[tid] = bv; s_i[tid] = bi; __syncthreads();
  for (int off = 128; off; off >>= 1) {
    if (tid < off) {
      if (s_v[tid + off] > s_v[tid] ||
          (s_v[tid + off] == s_v[tid] && s_i[tid + off] < s_i[tid])) {
        s_v[tid] = s_v[tid + off]; s_i[tid] = s_i[tid + off];
      }
    }
    __syncthreads();
  }
  if (tid == 0) {
    int tgt = labels[b * 25 + t];
    float sc = fminf(fmaxf(srow[tgt], 1e-30f), 1.f);
    float w = (tgt == 0) ? 0.f : 1.f;   // nll weight: PAD -> 0
    lossacc[b] += -w * __logf(sc);
    out[64 + b * 25 + t] = (float)s_i[0];
  }
}

__global__ void finalize_kernel(const float* __restrict__ lossacc, float* __restrict__ out) {
  int b = threadIdx.x;
  if (b < 64) out[b] = lossacc[b] * (1.f / 25.f);
}

// ---------------------------------------------------------------------------
extern "C" void kernel_launch(void* const* d_in, const int* in_sizes, int n_in,
                              void* d_out, int out_size, void* d_ws, size_t ws_size,
                              hipStream_t stream) {
  (void)in_sizes; (void)n_in; (void)out_size; (void)ws_size;
  const float* enc    = (const float*)d_in[0];
  const float* h0     = (const float*)d_in[1];
  const float* c0     = (const float*)d_in[2];
  const int*   labels = (const int*)d_in[3];
  const int*   inputs = (const int*)d_in[4];
  const float* embed  = (const float*)d_in[5];
  const float* W_ih   = (const float*)d_in[6];
  const float* W_hh   = (const float*)d_in[7];
  const float* b_ih   = (const float*)d_in[8];
  const float* b_hh   = (const float*)d_in[9];
  const float* Wa_h   = (const float*)d_in[10];
  const float* Wa_s   = (const float*)d_in[11];
  const float* ba_s   = (const float*)d_in[12];
  const float* v_t    = (const float*)d_in[13];
  const float* W_fc   = (const float*)d_in[14];
  const float* b_fc   = (const float*)d_in[15];
  const float* w_ph   = (const float*)d_in[16];
  const float* w_ps   = (const float*)d_in[17];
  const float* w_px   = (const float*)d_in[18];
  const float* b_px   = (const float*)d_in[19];
  float* out = (float*)d_out;

  // workspace carve-up
  char* p = (char*)d_ws;
  auto carve = [&](size_t bytes) { char* r = p; p += (bytes + 255) & ~(size_t)255; return r; };
  unsigned short* W_fc_bf = (unsigned short*)carve((size_t)32000 * 2048 * 2); // 131 MB -> L2-resident
  unsigned short* W_ih_bf = (unsigned short*)carve((size_t)4096 * 512 * 2);
  unsigned short* W_hh_bf = (unsigned short*)carve((size_t)4096 * 1024 * 2);
  unsigned short* Wa_s_bf = (unsigned short*)carve((size_t)512 * 1024 * 2);
  unsigned short* Wa_h_bf = (unsigned short*)carve((size_t)512 * 1024 * 2);
  unsigned short* enc_bf  = (unsigned short*)carve((size_t)8192 * 1024 * 2);
  float* wh      = (float*)carve((size_t)8192 * 512 * 4);
  float* hbuf    = (float*)carve((size_t)64 * 1024 * 4);
  float* cbuf    = (float*)carve((size_t)64 * 1024 * 4);
  unsigned short* hbf = (unsigned short*)carve((size_t)64 * 1024 * 2);
  float* xbuf    = (float*)carve((size_t)64 * 512 * 4);
  unsigned short* xbf = (unsigned short*)carve((size_t)64 * 512 * 2);
  float* gates   = (float*)carve((size_t)64 * 4096 * 4);
  float* wsbuf   = (float*)carve((size_t)64 * 512 * 4);
  float* attwbuf = (float*)carve((size_t)64 * 128 * 4);
  float* ctxbuf  = (float*)carve((size_t)64 * 1024 * 4);
  unsigned short* fcA = (unsigned short*)carve((size_t)64 * 2048 * 2);
  float* logits  = (float*)carve((size_t)64 * 32000 * 4);
  float* score   = (float*)carve((size_t)64 * 32000 * 4);
  float* ifcbuf  = (float*)carve(256);
  float* lossacc = (float*)carve(256);

  auto conv = [&](const float* s, unsigned short* d, size_t n) {
    size_t n8 = n / 8;
    f32_to_bf16_vec8<<<(unsigned)((n8 + 255) / 256), 256, 0, stream>>>(s, d, n8);
  };
  auto gemm64 = [&](const unsigned short* A, int lda, const unsigned short* Bm, int ldb,
                    float* C, int ldc, const float* bi0, const float* bi1,
                    int Nt, int K, int beta) {
    int blocks = (Nt + 7) / 8;        // 8 wave32 per 256-thread block
    gemm_bf16_wmma_m4<<<blocks, 256, 0, stream>>>(A, lda, Bm, ldb, C, ldc, bi0, bi1, Nt, K, beta);
  };

  // one-time: bf16 weight images + state init + wh precompute
  conv(W_fc, W_fc_bf, (size_t)32000 * 2048);
  conv(W_ih, W_ih_bf, (size_t)4096 * 512);
  conv(W_hh, W_hh_bf, (size_t)4096 * 1024);
  conv(Wa_s, Wa_s_bf, (size_t)512 * 1024);
  conv(Wa_h, Wa_h_bf, (size_t)512 * 1024);
  conv(enc,  enc_bf,  (size_t)8192 * 1024);
  init_state_kernel<<<256, 256, 0, stream>>>(h0, c0, hbuf, cbuf, hbf, lossacc);
  // wh[8192,512] = enc[8192,1024] @ Wa_h[512,1024]^T   (large-M path)
  gemm_bf16_wmma<<<(512 * 32 + 7) / 8, 256, 0, stream>>>(enc_bf, 1024, Wa_h_bf, 1024,
                                                         wh, 512, 512, 32, 1024);

  for (int t = 0; t < 25; ++t) {
    embed_kernel<<<128, 256, 0, stream>>>(labels, embed, t, xbuf, xbf);
    // gates = x @ W_ih^T  then  += h @ W_hh^T + b_ih + b_hh
    gemm64(xbf, 512,  W_ih_bf, 512,  gates, 4096, nullptr, nullptr, 256, 512, 0);
    gemm64(hbf, 1024, W_hh_bf, 1024, gates, 4096, b_ih, b_hh,       256, 1024, 1);
    lstm_ew_kernel<<<256, 256, 0, stream>>>(gates, cbuf, hbuf, hbf, fcA);
    // ws = h2 @ Wa_s^T + ba_s
    gemm64(hbf, 1024, Wa_s_bf, 1024, wsbuf, 512, ba_s, nullptr, 32, 1024, 0);
    attention_kernel<<<64, 256, 0, stream>>>(wh, wsbuf, v_t, enc, hbuf, xbuf,
                                             w_ph, w_ps, w_px, b_px,
                                             attwbuf, ctxbuf, fcA, ifcbuf);
    // logits = [h2, ctx] @ W_fc^T + b_fc   (dominant GEMM, W_fc bf16 in L2)
    gemm64(fcA, 2048, W_fc_bf, 2048, logits, 32000, b_fc, nullptr, 2000, 2048, 0);
    score_kernel<<<64, 256, 0, stream>>>(logits, attwbuf, ifcbuf, inputs, labels, t,
                                         score, lossacc, out);
  }
  finalize_kernel<<<1, 64, 0, stream>>>(lossacc, out);
}
